// AdapterExpert_78460462563328
// MI455X (gfx1250) — matrix operand order repaired and checked by reading, not compiled, not run on hardware
//
#include <hip/hip_runtime.h>
#include <math.h>

#define B_    32
#define IMG_  224
#define P_    16
#define CIN_  3
#define D_    768
#define H_    12
#define L_    12
#define A_    6
#define E_    8
#define DD_   64
#define FF_   3072
#define G_    14
#define T_    196      // patches per image
#define NV_   197      // valid tokens (cls + patches)
#define NP_   208      // padded tokens (13 * 16)
#define HD_   64

typedef __attribute__((ext_vector_type(16))) __bf16 bf16x16;
typedef __attribute__((ext_vector_type(8)))  __bf16 bf16x8;
typedef __attribute__((ext_vector_type(8)))  float  f32x8;

__device__ __forceinline__ unsigned int f2bf(float f) {
  union { float f; unsigned int u; } c; c.f = f;
  const unsigned int u = c.u;
  return (u + 0x7FFFu + ((u >> 16) & 1u)) >> 16;   // RNE, low 16 bits valid
}
__device__ __forceinline__ unsigned int pack2bf(float x, float y) {
  return (f2bf(x) & 0xFFFFu) | (f2bf(y) << 16);
}

__device__ __forceinline__ bf16x16 ld_frag(const unsigned short* p0, const unsigned short* p1) {
  bf16x8 lo = *(const bf16x8*)p0;   // 16B aligned -> ds_read_b128
  bf16x8 hi = *(const bf16x8*)p1;
  return __builtin_shufflevector(lo, hi, 0,1,2,3,4,5,6,7,8,9,10,11,12,13,14,15);
}

__device__ __forceinline__ float gelu_exact(float x) {
  return 0.5f * x * (1.0f + erff(x * 0.70710678118654752f));
}

// ---------------------------------------------------------------------------
// Generic batched GEMM: C[M,N] = A[M,K] * B[K,N] (+bias) (+gelu)
// BT=1: B stored as N x K row-major (i.e. B^T), ldb = K-dim row stride.
// Batch offset: z1 = z/batchDiv, z2 = z%batchDiv; off = z1*s1 + z2*s2.
// Optional expert gather: Bptr = Bw + bIdx[z]*idxStrB (bias likewise).
// Block tile 128x64, 8 waves (wave32), each wave 32x32 via 2x2 WMMA 16x16x32.
// Double-buffered LDS, register-staged global prefetch, one barrier / K-step.
// ---------------------------------------------------------------------------
#define BM 128
#define BN 64
#define BK 32
#define LSA 40
#define LSB 40

template<int BT, int EPI>
__global__ __launch_bounds__(256)
void gemm_wmma_kernel(const float* __restrict__ A, const float* __restrict__ Bw,
                      const float* __restrict__ bias, float* __restrict__ C,
                      int M, int N, int K, int lda, int ldb, int ldc,
                      long long sA1, long long sA2, long long sB1, long long sB2,
                      long long sC1, long long sC2, int batchDiv,
                      const int* __restrict__ bIdx, long long idxStrB, long long idxStrBias)
{
  __shared__ unsigned short As[2][BM * LSA];
  __shared__ unsigned short Bs[2][BN * LSB];

  const int z  = blockIdx.z;
  const int z1 = z / batchDiv;
  const int z2 = z - z1 * batchDiv;

  const float* Ab = A + z1 * sA1 + z2 * sA2;
  const float* Bb;
  const float* biasb = bias;
  if (bIdx) {
    const long long e = bIdx[z];
    Bb = Bw + e * idxStrB;
    if (bias) biasb = bias + e * idxStrBias;
  } else {
    Bb = Bw + z1 * sB1 + z2 * sB2;
  }
  float* Cb = C + z1 * sC1 + z2 * sC2;

  const int rowBase = blockIdx.y * BM;
  const int colBase = blockIdx.x * BN;
  const int tid  = threadIdx.x;
  const int lane = tid & 31;
  const int wv   = tid >> 5;
  const int wm   = wv >> 1;       // 0..3 (M direction)
  const int wn   = wv & 1;        // 0..1 (N direction)
  const int l16  = lane & 15;
  const int half = lane >> 4;     // 0/1

  // A fill geometry: 2048 float2 pairs; thread handles 8 (row = p>>4, col = (p&15)*2)
  const int aRow = tid >> 4;            // base row for r=0 chunk; +16 rows per chunk
  const int aCol = (tid & 15) << 1;
  // BT fill: 1024 pairs; thread handles 4 (n = p>>4, col likewise)
  // non-BT fill: 1024 float2 (along n); thread handles 4 (k = p>>5, n = (p&31)*2)
  const int bKk = tid >> 5;
  const int bN2 = (tid & 31) << 1;

  const bool aRowsIn = (rowBase + BM <= M);
  const bool bColsIn = (colBase + BN <= N);

  float2 ra[8];
  float2 rb[4];

  auto loadA = [&](int k0) {
    if (aRowsIn && (k0 + BK <= K)) {
      #pragma unroll
      for (int r = 0; r < 8; ++r)
        ra[r] = *(const float2*)(Ab + (long long)(rowBase + aRow + r * 16) * lda + (k0 + aCol));
    } else {
      #pragma unroll
      for (int r = 0; r < 8; ++r) {
        float2 v = make_float2(0.f, 0.f);
        const int gr = rowBase + aRow + r * 16, gk = k0 + aCol;
        if (gr < M && gk < K) v = *(const float2*)(Ab + (long long)gr * lda + gk);
        ra[r] = v;
      }
    }
  };
  auto loadB = [&](int k0) {
    if (BT) {
      if (bColsIn && (k0 + BK <= K)) {
        #pragma unroll
        for (int r = 0; r < 4; ++r)
          rb[r] = *(const float2*)(Bb + (long long)(colBase + aRow + r * 16) * ldb + (k0 + aCol));
      } else {
        #pragma unroll
        for (int r = 0; r < 4; ++r) {
          float2 v = make_float2(0.f, 0.f);
          const int gn = colBase + aRow + r * 16, gk = k0 + aCol;
          if (gn < N && gk < K) v = *(const float2*)(Bb + (long long)gn * ldb + gk);
          rb[r] = v;
        }
      }
    } else {
      if (bColsIn && (k0 + BK <= K)) {
        #pragma unroll
        for (int r = 0; r < 4; ++r)
          rb[r] = *(const float2*)(Bb + (long long)(k0 + bKk + r * 8) * ldb + (colBase + bN2));
      } else {
        #pragma unroll
        for (int r = 0; r < 4; ++r) {
          float2 v = make_float2(0.f, 0.f);
          const int gk = k0 + bKk + r * 8, gn = colBase + bN2;
          if (gk < K && gn < N) v = *(const float2*)(Bb + (long long)gk * ldb + gn);
          rb[r] = v;
        }
      }
    }
  };
  auto stage = [&](int cur) {
    unsigned short* as = As[cur];
    unsigned short* bs = Bs[cur];
    #pragma unroll
    for (int r = 0; r < 8; ++r)
      *(unsigned int*)&as[(aRow + r * 16) * LSA + aCol] = pack2bf(ra[r].x, ra[r].y);
    if (BT) {
      #pragma unroll
      for (int r = 0; r < 4; ++r)
        *(unsigned int*)&bs[(aRow + r * 16) * LSB + aCol] = pack2bf(rb[r].x, rb[r].y);
    } else {
      #pragma unroll
      for (int r = 0; r < 4; ++r) {
        const int kk = bKk + r * 8;
        bs[bN2 * LSB + kk]       = (unsigned short)f2bf(rb[r].x);
        bs[(bN2 + 1) * LSB + kk] = (unsigned short)f2bf(rb[r].y);
      }
    }
  };

  f32x8 acc[2][2] = {};

  loadA(0);
  loadB(0);
  int cur = 0;
  for (int k0 = 0; k0 < K; k0 += BK) {
    stage(cur);
    const int k1 = k0 + BK;
    if (k1 < K) { loadA(k1); loadB(k1); }   // in flight during barrier + wmma
    __syncthreads();

    bf16x16 af[2], bfr[2];
    #pragma unroll
    for (int i = 0; i < 2; ++i) {
      // A frag: row = M (lane&15), half selects K base 0/8; [kb,kb+8) then [kb+16,kb+24)
      const unsigned short* pa = &As[cur][(wm * 32 + i * 16 + l16) * LSA + half * 8];
      af[i] = ld_frag(pa, pa + 16);
    }
    #pragma unroll
    for (int j = 0; j < 2; ++j) {
      // B frag: col = N (lane&15), half selects K base 0/8; [kb,kb+8) then [kb+16,kb+24)
      const unsigned short* pb = &Bs[cur][(wn * 32 + j * 16 + l16) * LSB + half * 8];
      bfr[j] = ld_frag(pb, pb + 16);
    }
    #pragma unroll
    for (int i = 0; i < 2; ++i)
      #pragma unroll
      for (int j = 0; j < 2; ++j)
        acc[i][j] = __builtin_amdgcn_wmma_f32_16x16x32_bf16(
            false, af[i], false, bfr[j], (short)0, acc[i][j], false, false);
    cur ^= 1;
  }

  // ---- epilogue: C layout lanes0-15 rows 0..7, lanes16-31 rows 8..15
  const bool interiorC = aRowsIn && bColsIn;
  #pragma unroll
  for (int i = 0; i < 2; ++i) {
    const int r0 = rowBase + wm * 32 + i * 16 + half * 8;
    #pragma unroll
    for (int j = 0; j < 2; ++j) {
      const int c = colBase + wn * 32 + j * 16 + l16;
      if (!interiorC && c >= N) continue;
      float bv = 0.f;
      if (EPI >= 1 && biasb) bv = biasb[c];
      #pragma unroll
      for (int v8 = 0; v8 < 8; ++v8) {
        const int rr = r0 + v8;
        if (interiorC || rr < M) {
          float out = acc[i][j][v8] + bv;
          if (EPI == 2) out = gelu_exact(out);
          Cb[(long long)rr * ldc + c] = out;
        }
      }
    }
  }
}

// ---------------------------------------------------------------------------
// LayerNorm over D columns; one block per row.
// ---------------------------------------------------------------------------
__global__ __launch_bounds__(256)
void ln_kernel(const float* __restrict__ x, const float* __restrict__ g,
               const float* __restrict__ b, float* __restrict__ y,
               long long inStride, long long outStride, int D)
{
  __shared__ float red[256];
  const long long r = blockIdx.x;
  const float* xr = x + r * inStride;
  float s = 0.f;
  for (int c = threadIdx.x; c < D; c += 256) s += xr[c];
  red[threadIdx.x] = s; __syncthreads();
  for (int o = 128; o > 0; o >>= 1) { if (threadIdx.x < o) red[threadIdx.x] += red[threadIdx.x + o]; __syncthreads(); }
  const float mean = red[0] / (float)D; __syncthreads();
  float s2 = 0.f;
  for (int c = threadIdx.x; c < D; c += 256) { float d = xr[c] - mean; s2 += d * d; }
  red[threadIdx.x] = s2; __syncthreads();
  for (int o = 128; o > 0; o >>= 1) { if (threadIdx.x < o) red[threadIdx.x] += red[threadIdx.x + o]; __syncthreads(); }
  const float inv = rsqrtf(red[0] / (float)D + 1e-6f);
  float* yr = y + r * outStride;
  for (int c = threadIdx.x; c < D; c += 256) yr[c] = (xr[c] - mean) * inv * g[c] + b[c];
}

// Softmax over one row of `cols` scores; columns >= valid masked out; scale applied.
__global__ __launch_bounds__(256)
void softmax_kernel(float* __restrict__ s, int cols, int valid, float scale)
{
  __shared__ float red[256];
  float* sr = s + (long long)blockIdx.x * cols;
  const int c = threadIdx.x;
  float v = -3.0e38f;
  if (c < valid) v = sr[c] * scale;
  red[c] = v; __syncthreads();
  for (int o = 128; o > 0; o >>= 1) { if (c < o) red[c] = fmaxf(red[c], red[c + o]); __syncthreads(); }
  const float mx = red[0]; __syncthreads();
  float e = 0.f;
  if (c < valid) e = expf(sr[c] * scale - mx);
  red[c] = e; __syncthreads();
  for (int o = 128; o > 0; o >>= 1) { if (c < o) red[c] += red[c + o]; __syncthreads(); }
  const float inv = 1.f / red[0];
  if (c < cols) sr[c] = (c < valid) ? e * inv : 0.f;
}

// im2col: xp[b, t, c*256 + p1*16 + p2] = in[b, c, g1*16+p1, g2*16+p2]
__global__ void im2col_kernel(const float* __restrict__ in, float* __restrict__ xp, long long total)
{
  long long i = (long long)blockIdx.x * blockDim.x + threadIdx.x;
  if (i >= total) return;
  const int d  = (int)(i % (CIN_ * P_ * P_));
  const long long rr = i / (CIN_ * P_ * P_);
  const int t  = (int)(rr % T_);
  const int b  = (int)(rr / T_);
  const int c  = d / (P_ * P_);
  const int pp = d % (P_ * P_);
  const int p1 = pp / P_, p2 = pp % P_;
  const int g1 = t / G_,  g2 = t % G_;
  xp[i] = in[(((long long)b * CIN_ + c) * IMG_ + (g1 * P_ + p1)) * IMG_ + (g2 * P_ + p2)];
}

// x[b,n,d] = n==0 ? cls+pos : (n<197 ? patch+pos : 0)
__global__ void assemble_kernel(const float* __restrict__ pe, const float* __restrict__ cls,
                                const float* __restrict__ pos, float* __restrict__ x, long long total)
{
  long long i = (long long)blockIdx.x * blockDim.x + threadIdx.x;
  if (i >= total) return;
  const int d = (int)(i % D_);
  const long long rr = i / D_;
  const int n = (int)(rr % NP_);
  const int b = (int)(rr / NP_);
  float v;
  if (n == 0)       v = cls[d] + pos[d];
  else if (n < NV_) v = pe[((long long)b * T_ + (n - 1)) * D_ + d] + pos[(long long)n * D_ + d];
  else              v = 0.f;
  x[i] = v;
}

__global__ void add_scaled_kernel(float* __restrict__ x, const float* __restrict__ a,
                                  const float* __restrict__ g, long long n, int D)
{
  long long i = (long long)blockIdx.x * blockDim.x + threadIdx.x;
  if (i < n) x[i] += a[i] * g[(int)(i % D)];
}

__global__ void add_kernel(float* __restrict__ x, const float* __restrict__ a, long long n)
{
  long long i = (long long)blockIdx.x * blockDim.x + threadIdx.x;
  if (i < n) x[i] += a[i];
}

// ---------------------------------------------------------------------------
extern "C" void kernel_launch(void* const* d_in, const int* in_sizes, int n_in,
                              void* d_out, int out_size, void* d_ws, size_t ws_size,
                              hipStream_t stream)
{
  (void)in_sizes; (void)n_in; (void)out_size; (void)ws_size;
  const float* inputs  = (const float*)d_in[0];
  const int*   expert  = (const int*)d_in[1];
  const float* patch_w = (const float*)d_in[2];
  const float* patch_b = (const float*)d_in[3];
  const float* cls_tok = (const float*)d_in[4];
  const float* pos_emb = (const float*)d_in[5];
  const float* ln1_g   = (const float*)d_in[6];
  const float* ln1_b   = (const float*)d_in[7];
  const float* qkv_w   = (const float*)d_in[8];
  const float* qkv_b   = (const float*)d_in[9];
  const float* proj_w  = (const float*)d_in[10];
  const float* proj_b  = (const float*)d_in[11];
  const float* ls1     = (const float*)d_in[12];
  const float* ln2_g   = (const float*)d_in[13];
  const float* ln2_b   = (const float*)d_in[14];
  const float* fc1_w   = (const float*)d_in[15];
  const float* fc1_b   = (const float*)d_in[16];
  const float* fc2_w   = (const float*)d_in[17];
  const float* fc2_b   = (const float*)d_in[18];
  const float* ls2     = (const float*)d_in[19];
  const float* norm_g  = (const float*)d_in[20];
  const float* norm_b  = (const float*)d_in[21];
  const float* adw     = (const float*)d_in[22];
  const float* adb     = (const float*)d_in[23];
  const float* auw     = (const float*)d_in[24];
  const float* aub     = (const float*)d_in[25];

  const long long NB = (long long)B_ * NP_;           // 6656 rows
  float* w    = (float*)d_ws;
  float* x    = w;  w += NB * D_;
  float* xln  = w;  w += NB * D_;                      // also proj-out / mlp-out
  float* qkv  = w;  w += NB * 3 * D_;                  // also patch-embed out
  float* attn = w;  w += NB * D_;
  float* hid  = w;  w += NB * FF_;                     // also scores / im2col
  float* adpt = w;  w += NB * D_;
  float* adh  = w;  /* w += NB * DD_; */
  float* scores = hid;
  float* xp     = hid;
  float* pbuf   = qkv;

  const dim3 blk(256);

  // ---- patch embedding -----------------------------------------------------
  {
    const long long tot = (long long)B_ * T_ * D_;
    im2col_kernel<<<dim3((unsigned)((tot + 255) / 256)), blk, 0, stream>>>(inputs, xp, tot);
    // pbuf = xp @ patch_w^T + patch_b   (patch_w is N x K row-major -> BT)
    gemm_wmma_kernel<1, 1><<<dim3(D_ / BN, (B_ * T_) / BM, 1), blk, 0, stream>>>(
        xp, patch_w, patch_b, pbuf,
        B_ * T_, D_, D_, D_, D_, D_,
        0, 0, 0, 0, 0, 0, 1, nullptr, 0, 0);
    const long long tot2 = NB * D_;
    assemble_kernel<<<dim3((unsigned)((tot2 + 255) / 256)), blk, 0, stream>>>(pbuf, cls_tok, pos_emb, x, tot2);
  }

  const long long nXD = NB * D_;
  const unsigned gEl  = (unsigned)((nXD + 255) / 256);

  for (int l = 0; l < L_; ++l) {
    const float* qw = qkv_w + (long long)l * D_ * 3 * D_;
    const float* qb = qkv_b + (long long)l * 3 * D_;
    const float* pw = proj_w + (long long)l * D_ * D_;
    const float* pb = proj_b + (long long)l * D_;
    const float* f1w = fc1_w + (long long)l * D_ * FF_;
    const float* f1b = fc1_b + (long long)l * FF_;
    const float* f2w = fc2_w + (long long)l * FF_ * D_;
    const float* f2b = fc2_b + (long long)l * D_;

    // LN1
    ln_kernel<<<dim3((unsigned)NB), blk, 0, stream>>>(x, ln1_g + l * D_, ln1_b + l * D_, xln, D_, D_, D_);
    // QKV: (NB x 768) @ (768 x 2304) + bias
    gemm_wmma_kernel<0, 1><<<dim3(3 * D_ / BN, (unsigned)(NB / BM), 1), blk, 0, stream>>>(
        xln, qw, qb, qkv, (int)NB, 3 * D_, D_, D_, 3 * D_, 3 * D_,
        0, 0, 0, 0, 0, 0, 1, nullptr, 0, 0);
    // scores[z=(b,h)] = Q @ K^T   (batched, BT: K stored tokens x hd)
    gemm_wmma_kernel<1, 0><<<dim3((NP_ + BN - 1) / BN, (NP_ + BM - 1) / BM, B_ * H_), blk, 0, stream>>>(
        qkv, qkv + D_, nullptr, scores,
        NP_, NP_, HD_, 3 * D_, 3 * D_, NP_,
        (long long)NP_ * 3 * D_, 64, (long long)NP_ * 3 * D_, 64,
        (long long)H_ * NP_ * NP_, (long long)NP_ * NP_,
        H_, nullptr, 0, 0);
    // softmax rows (mask cols >= 197, scale 1/8)
    softmax_kernel<<<dim3((unsigned)(B_ * H_ * NP_)), blk, 0, stream>>>(scores, NP_, NV_, 0.125f);
    // attn[z] = P @ V  -> laid out (b, n, h*64+d)
    gemm_wmma_kernel<0, 0><<<dim3(1, (NP_ + BM - 1) / BM, B_ * H_), blk, 0, stream>>>(
        scores, qkv + 2 * D_, nullptr, attn,
        NP_, HD_, NP_, NP_, 3 * D_, D_,
        (long long)H_ * NP_ * NP_, (long long)NP_ * NP_,
        (long long)NP_ * 3 * D_, 64,
        (long long)NP_ * D_, 64,
        H_, nullptr, 0, 0);
    // proj (-> xln reused as scratch) then x += proj * ls1
    gemm_wmma_kernel<0, 1><<<dim3(D_ / BN, (unsigned)(NB / BM), 1), blk, 0, stream>>>(
        attn, pw, pb, xln, (int)NB, D_, D_, D_, D_, D_,
        0, 0, 0, 0, 0, 0, 1, nullptr, 0, 0);
    add_scaled_kernel<<<dim3(gEl), blk, 0, stream>>>(x, xln, ls1 + l * D_, nXD, D_);

    // adapter (layers 0..5): h = gelu(x @ dw[e] + db[e]); adpt = h @ uw[e] + ub[e]
    if (l < A_) {
      gemm_wmma_kernel<0, 2><<<dim3(1, (NP_ + BM - 1) / BM, B_), blk, 0, stream>>>(
          x, adw + (long long)l * E_ * D_ * DD_, adb + (long long)l * E_ * DD_, adh,
          NP_, DD_, D_, D_, DD_, DD_,
          (long long)NP_ * D_, 0, 0, 0, (long long)NP_ * DD_, 0,
          1, expert, (long long)D_ * DD_, DD_);
      gemm_wmma_kernel<0, 1><<<dim3(D_ / BN, (NP_ + BM - 1) / BM, B_), blk, 0, stream>>>(
          adh, auw + (long long)l * E_ * DD_ * D_, aub + (long long)l * E_ * D_, adpt,
          NP_, D_, DD_, DD_, D_, D_,
          (long long)NP_ * DD_, 0, 0, 0, (long long)NP_ * D_, 0,
          1, expert, (long long)DD_ * D_, D_);
    }

    // LN2 -> xln; MLP: hid = gelu(xln@fc1+b); xln = hid@fc2+b
    ln_kernel<<<dim3((unsigned)NB), blk, 0, stream>>>(x, ln2_g + l * D_, ln2_b + l * D_, xln, D_, D_, D_);
    gemm_wmma_kernel<0, 2><<<dim3(FF_ / BN, (unsigned)(NB / BM), 1), blk, 0, stream>>>(
        xln, f1w, f1b, hid, (int)NB, FF_, D_, D_, FF_, FF_,
        0, 0, 0, 0, 0, 0, 1, nullptr, 0, 0);
    gemm_wmma_kernel<0, 1><<<dim3(D_ / BN, (unsigned)(NB / BM), 1), blk, 0, stream>>>(
        hid, f2w, f2b, xln, (int)NB, D_, FF_, FF_, D_, D_,
        0, 0, 0, 0, 0, 0, 1, nullptr, 0, 0);

    if (l < A_) add_kernel<<<dim3(gEl), blk, 0, stream>>>(x, adpt, nXD);
    add_scaled_kernel<<<dim3(gEl), blk, 0, stream>>>(x, xln, ls2 + l * D_, nXD, D_);
  }

  // final LN on token 0 of each image -> d_out (B x 768)
  ln_kernel<<<dim3(B_), blk, 0, stream>>>(x, norm_g, norm_b, (float*)d_out,
                                          (long long)NP_ * D_, D_, D_);
}
